// AttentionDecoderCell_74208444940657
// MI455X (gfx1250) — matrix operand order repaired and checked by reading, not compiled, not run on hardware
//
#include <hip/hip_runtime.h>
#include <hip/hip_bf16.h>

#define B_ 64
#define T_ 2048
#define E_ 256
#define U_ 256
#define D_ 256
#define O_ 32000
#define C_ 512   // U+E

typedef __attribute__((ext_vector_type(16))) __bf16 v16bf;
typedef __attribute__((ext_vector_type(8)))  float  v8f;
typedef __attribute__((ext_vector_type(4)))  unsigned u32x4;
typedef __attribute__((ext_vector_type(4)))  int      i32x4;
typedef __attribute__((ext_vector_type(8)))  int      i32x8;

union BF16x16 { v16bf v; unsigned u[8]; };

__device__ __forceinline__ unsigned short f2bf(float f) {
    unsigned u = __float_as_uint(f);
    u += 0x7FFFu + ((u >> 16) & 1u);   // round-to-nearest-even
    return (unsigned short)(u >> 16);
}
__device__ __forceinline__ unsigned pack2(float lo, float hi) {
    return (unsigned)f2bf(lo) | ((unsigned)f2bf(hi) << 16);
}
// branch-free tanh: 1 - 2/(e^{2x}+1); saturates via inf/0 in __expf.
__device__ __forceinline__ float fast_tanh(float x) {
    float e = __expf(2.0f * x);
    return 1.0f - 2.0f * __builtin_amdgcn_rcpf(e + 1.0f);
}

// ---------------------------------------------------------------------------
// TDM: issue a 2D tensor_load_to_lds (contiguous rows of f32).
// D# per CDNA5 ISA; 6-arg builtin (clang-23).
// ---------------------------------------------------------------------------
__device__ __forceinline__ void tdm_load_2d_f32(unsigned lds_off, const void* gaddr,
                                                unsigned tile_rows, unsigned row_elems,
                                                unsigned row_stride) {
    unsigned long long ga = (unsigned long long)gaddr;
    u32x4 g0;
    g0[0] = 1u;                                             // count=1 (valid user D#)
    g0[1] = lds_off;                                        // LDS byte address
    g0[2] = (unsigned)(ga & 0xFFFFFFFFull);                 // global_addr[31:0]
    g0[3] = (unsigned)((ga >> 32) & 0x01FFFFFFull)          // global_addr[56:32]
          | (2u << 30);                                     // type=2 (image)
    i32x8 g1;
    g1[0] = (int)(2u << 16);                                // data_size=2 -> 4 bytes
    g1[1] = (int)(row_elems << 16);                         // tensor_dim0[15:0] in [31:16]
    g1[2] = (int)((row_elems >> 16) | (tile_rows << 16));   // dim0 hi | tensor_dim1 lo
    g1[3] = (int)(row_elems << 16);                         // dim1 hi(0) | tile_dim0
    g1[4] = (int)tile_rows;                                 // tile_dim1 | tile_dim2=0
    g1[5] = (int)row_stride;                                // tensor_dim0_stride[31:0]
    g1[6] = 0;
    g1[7] = 0;
    i32x4 gz;  gz[0] = 0; gz[1] = 0; gz[2] = 0; gz[3] = 0;  // groups 2/3 unused (2D)
    i32x8 gz8; gz8[0] = 0; gz8[1] = 0; gz8[2] = 0; gz8[3] = 0;
               gz8[4] = 0; gz8[5] = 0; gz8[6] = 0; gz8[7] = 0;
    __builtin_amdgcn_tensor_load_to_lds(g0, g1, gz, gz, gz8, 0);
}

// ---------------------------------------------------------------------------
// 1) Pre-swizzle W_a[256:512, :] into per-lane bf16 B-frags.
// ---------------------------------------------------------------------------
__global__ void __launch_bounds__(256) k_prep_wax(const float* __restrict__ W_a,
                                                  unsigned* __restrict__ WaxSw) {
    int idx  = blockIdx.x * 256 + threadIdx.x;          // 65536 total
    int j    = idx & 7;
    int lane = (idx >> 3) & 31;
    int kt   = (idx >> 8) & 7;
    int nt   = idx >> 11;
    int n    = nt * 16 + (lane & 15);
    int k0   = kt * 32 + (lane >> 4) * 8 + (j >> 2) * 16 + 2 * (j & 3);      // 0..254
    float f0 = W_a[(size_t)(256 + k0)     * C_ + n];
    float f1 = W_a[(size_t)(256 + k0 + 1) * C_ + n];
    WaxSw[idx] = pack2(f0, f1);
}

// ---------------------------------------------------------------------------
// 2) stm_part[b][n] = stm[b,:] @ W_a[0:256, n]
// ---------------------------------------------------------------------------
__global__ void __launch_bounds__(256) k_stm_part(const float* __restrict__ stm,
                                                  const float* __restrict__ W_a,
                                                  float* __restrict__ stm_part) {
    int idx = blockIdx.x * 256 + threadIdx.x;           // 64*512
    int b = idx >> 9, n = idx & 511;
    const float* s = stm + (size_t)b * U_;
    float acc = 0.f;
    #pragma unroll 4
    for (int k = 0; k < U_; ++k) acc += s[k] * W_a[(size_t)k * C_ + n];
    stm_part[idx] = acc;
}

// ---------------------------------------------------------------------------
// 3) Attention scores: one wave per 16 (b,t) rows, bf16 WMMA.
// ---------------------------------------------------------------------------
__global__ void __launch_bounds__(32) k_scores(const float* __restrict__ x_seq,
                                               const unsigned* __restrict__ WaxSw,
                                               const float* __restrict__ stm_part,
                                               const float* __restrict__ V_a,
                                               float* __restrict__ et) {
    const int lane = threadIdx.x;
    const int m    = lane & 15;
    const int half = lane >> 4;
    const int koff = half * 8;
    const int tile = blockIdx.x;            // 0..8191
    const int b    = tile >> 7;
    const int t0   = (tile & 127) << 4;

    const float* xrow = x_seq + ((size_t)b * T_ + t0 + m) * E_;
    BF16x16 afr[8];
    #pragma unroll
    for (int kt = 0; kt < 8; ++kt) {
        const float* xp  = xrow + kt * 32 + koff;
        const float* xp2 = xp + 16;
        #pragma unroll
        for (int j = 0; j < 4; ++j) {
            afr[kt].u[j]     = pack2(xp[2*j],  xp[2*j+1]);
            afr[kt].u[4 + j] = pack2(xp2[2*j], xp2[2*j+1]);
        }
    }

    float psum[8];
    #pragma unroll
    for (int r = 0; r < 8; ++r) psum[r] = 0.f;
    const float* sp = stm_part + (size_t)b * C_;

    for (int nt = 0; nt < 32; ++nt) {
        v8f acc;
        #pragma unroll
        for (int r = 0; r < 8; ++r) acc[r] = 0.f;
        #pragma unroll
        for (int kt = 0; kt < 8; ++kt) {
            BF16x16 bfr;
            const unsigned* bp = WaxSw + ((((nt * 8 + kt) * 32) + lane) << 3);
            #pragma unroll
            for (int j = 0; j < 8; ++j) bfr.u[j] = bp[j];
            acc = __builtin_amdgcn_wmma_f32_16x16x32_bf16(
                false, afr[kt].v, false, bfr.v, (short)0, acc, false, false);
        }
        int   n   = nt * 16 + m;
        float va  = V_a[n];
        float spn = sp[n];
        #pragma unroll
        for (int r = 0; r < 8; ++r) psum[r] += fast_tanh(acc[r] + spn) * va;
    }

    #pragma unroll
    for (int r = 0; r < 8; ++r) {
        float v = psum[r];
        v += __shfl_xor(v, 1, 32);
        v += __shfl_xor(v, 2, 32);
        v += __shfl_xor(v, 4, 32);
        v += __shfl_xor(v, 8, 32);
        psum[r] = v;
    }
    if (m == 0) {
        #pragma unroll
        for (int r = 0; r < 8; ++r)
            et[(size_t)b * T_ + t0 + half * 8 + r] = psum[r];
    }
}

// ---------------------------------------------------------------------------
// 4) softmax over T + context, x_seq streamed via TDM double-buffer.
// ---------------------------------------------------------------------------
#define CTX_ROWS 32
__global__ void __launch_bounds__(256) k_softmax_ctx(const float* __restrict__ et,
                                                     const float* __restrict__ x_seq,
                                                     float* __restrict__ context) {
    __shared__ float s_at[T_];
    __shared__ float s_red[256];
    __shared__ float s_x[2][CTX_ROWS * E_];   // 2 x 32KB double buffer
    int b = blockIdx.x, tid = threadIdx.x;
    const float* e = et + (size_t)b * T_;

    float mx = -3.4e38f;
    for (int t = tid; t < T_; t += 256) mx = fmaxf(mx, e[t]);
    s_red[tid] = mx; __syncthreads();
    for (int s = 128; s > 0; s >>= 1) {
        if (tid < s) s_red[tid] = fmaxf(s_red[tid], s_red[tid + s]);
        __syncthreads();
    }
    mx = s_red[0]; __syncthreads();

    float sum = 0.f;
    for (int t = tid; t < T_; t += 256) {
        float v = __expf(e[t] - mx);
        s_at[t] = v;
        sum += v;
    }
    s_red[tid] = sum; __syncthreads();
    for (int s = 128; s > 0; s >>= 1) {
        if (tid < s) s_red[tid] += s_red[tid + s];
        __syncthreads();
    }
    float inv = 1.f / s_red[0];

    const float* xb = x_seq + (size_t)b * T_ * E_;
    const bool issuer = (tid < 32);          // wave 0 drives the TDM
    if (issuer)
        tdm_load_2d_f32((unsigned)(size_t)&s_x[0][0], xb, CTX_ROWS, E_, E_);

    float acc = 0.f;
    for (int tile = 0; tile < T_ / CTX_ROWS; ++tile) {
        int cur = tile & 1;
        if (issuer) {
            if (tile + 1 < T_ / CTX_ROWS) {
                tdm_load_2d_f32((unsigned)(size_t)&s_x[cur ^ 1][0],
                                xb + (size_t)(tile + 1) * CTX_ROWS * E_,
                                CTX_ROWS, E_, E_);
                __builtin_amdgcn_s_wait_tensorcnt(1);   // current tile landed
            } else {
                __builtin_amdgcn_s_wait_tensorcnt(0);
            }
        }
        __syncthreads();                                 // tile visible to all waves
        const float* bufp = &s_x[cur][0];
        const float* atp  = s_at + tile * CTX_ROWS;
        #pragma unroll
        for (int i = 0; i < CTX_ROWS; ++i)
            acc += atp[i] * bufp[i * E_ + tid];
        __syncthreads();                                 // done before buffer reuse
    }
    context[b * E_ + tid] = acc * inv;
}

// ---------------------------------------------------------------------------
// 5) GRU gates r, z
// ---------------------------------------------------------------------------
__global__ void __launch_bounds__(256) k_gates_rz(const float* __restrict__ inputs,
                                                  const float* __restrict__ stm,
                                                  const float* __restrict__ context,
                                                  const float* __restrict__ W_r, const float* __restrict__ U_r,
                                                  const float* __restrict__ C_r, const float* __restrict__ b_r,
                                                  const float* __restrict__ W_z, const float* __restrict__ U_z,
                                                  const float* __restrict__ C_z, const float* __restrict__ b_z,
                                                  float* __restrict__ rt, float* __restrict__ zt) {
    int b = blockIdx.x, u = threadIdx.x;
    const float* in = inputs  + (size_t)b * D_;
    const float* st = stm     + (size_t)b * U_;
    const float* cx = context + (size_t)b * E_;
    float ar = b_r[u], az = b_z[u];
    #pragma unroll 4
    for (int k = 0; k < 256; ++k) {
        float i = in[k], s = st[k], c = cx[k];
        ar += i * W_r[(size_t)k * U_ + u] + s * U_r[(size_t)k * U_ + u] + c * C_r[(size_t)k * U_ + u];
        az += i * W_z[(size_t)k * U_ + u] + s * U_z[(size_t)k * U_ + u] + c * C_z[(size_t)k * U_ + u];
    }
    rt[b * U_ + u] = 1.f / (1.f + __expf(-ar));
    zt[b * U_ + u] = 1.f / (1.f + __expf(-az));
}

// ---------------------------------------------------------------------------
// 6) s_tp + new state st  -> d_out tail (precise tanh here)
// ---------------------------------------------------------------------------
__global__ void __launch_bounds__(256) k_gates_st(const float* __restrict__ inputs,
                                                  const float* __restrict__ stm,
                                                  const float* __restrict__ context,
                                                  const float* __restrict__ rt, const float* __restrict__ zt,
                                                  const float* __restrict__ W_p, const float* __restrict__ U_p,
                                                  const float* __restrict__ C_p, const float* __restrict__ b_p,
                                                  float* __restrict__ st_out) {
    int b = blockIdx.x, u = threadIdx.x;
    const float* in = inputs  + (size_t)b * D_;
    const float* st = stm     + (size_t)b * U_;
    const float* cx = context + (size_t)b * E_;
    const float* r  = rt      + (size_t)b * U_;
    float ap = b_p[u];
    #pragma unroll 4
    for (int k = 0; k < 256; ++k) {
        ap += in[k] * W_p[(size_t)k * U_ + u]
            + (r[k] * st[k]) * U_p[(size_t)k * U_ + u]
            + cx[k] * C_p[(size_t)k * U_ + u];
    }
    float s_tp = tanhf(ap);
    float z    = zt[b * U_ + u];
    st_out[b * U_ + u] = (1.f - z) * st[u] + z * s_tp;
}

// ---------------------------------------------------------------------------
// 7) bf16 activations for the output GEMM
// ---------------------------------------------------------------------------
__global__ void __launch_bounds__(256) k_actbf(const float* __restrict__ inputs,
                                               const float* __restrict__ stm,
                                               const float* __restrict__ context,
                                               unsigned short* __restrict__ act) {
    int b = blockIdx.x, tid = threadIdx.x;
    act[(size_t)b * 768 + tid]       = f2bf(inputs[b * D_ + tid]);
    act[(size_t)b * 768 + 256 + tid] = f2bf(stm[b * U_ + tid]);
    act[(size_t)b * 768 + 512 + tid] = f2bf(context[b * E_ + tid]);
}

// ---------------------------------------------------------------------------
// 8) Output logits: [64,768] @ [768,32000] + b_o, bf16 WMMA.
// ---------------------------------------------------------------------------
__global__ void __launch_bounds__(32) k_outgemm(const unsigned short* __restrict__ act,
                                                const float* __restrict__ W_o,
                                                const float* __restrict__ U_o,
                                                const float* __restrict__ C_o,
                                                const float* __restrict__ b_o,
                                                float* __restrict__ out) {
    const int lane = threadIdx.x;
    const int m16  = lane & 15;
    const int half = lane >> 4;
    const int koff = half * 16;
    const int n0   = blockIdx.x * 16;
    const int n    = n0 + m16;

    v8f acc[4];
    #pragma unroll
    for (int mt = 0; mt < 4; ++mt)
        #pragma unroll
        for (int r = 0; r < 8; ++r) acc[mt][r] = 0.f;

    for (int kt = 0; kt < 24; ++kt) {
        const float* W = (kt < 8) ? W_o : (kt < 16) ? U_o : C_o;
        int krel = (kt & 7) * 32 + half * 16;
        const float* wp = W + (size_t)krel * O_ + n;
        if ((kt & 7) != 7) __builtin_prefetch(wp + (size_t)32 * O_, 0, 0);

        BF16x16 bfr;
        #pragma unroll
        for (int j = 0; j < 8; ++j)
            bfr.u[j] = pack2(wp[(size_t)(2 * j) * O_], wp[(size_t)(2 * j + 1) * O_]);

        #pragma unroll
        for (int mt = 0; mt < 4; ++mt) {
            const unsigned* ap  = (const unsigned*)(act + (size_t)(mt * 16 + m16) * 768 + kt * 32 + koff);
            const unsigned* ap2 = (const unsigned*)(act + (size_t)(mt * 16 + m16) * 768 + kt * 32 + 8 + koff);
            BF16x16 afr;
            #pragma unroll
            for (int j = 0; j < 4; ++j) { afr.u[j] = ap[j]; afr.u[4 + j] = ap2[j]; }
            acc[mt] = __builtin_amdgcn_wmma_f32_16x16x32_bf16(
                false, afr.v, false, bfr.v, (short)0, acc[mt], false, false);
        }
    }

    float bias = b_o[n];
    #pragma unroll
    for (int mt = 0; mt < 4; ++mt)
        #pragma unroll
        for (int r = 0; r < 8; ++r)
            out[(size_t)(mt * 16 + half * 8 + r) * O_ + n] = acc[mt][r] + bias;
}

// ---------------------------------------------------------------------------
// 9) row softmax over O=32000, in place on d_out
// ---------------------------------------------------------------------------
__global__ void __launch_bounds__(256) k_outsoftmax(float* __restrict__ out) {
    __shared__ float s_red[256];
    int b = blockIdx.x, tid = threadIdx.x;
    float* row = out + (size_t)b * O_;

    float mx = -3.4e38f;
    for (int i = tid; i < O_; i += 256) mx = fmaxf(mx, row[i]);
    s_red[tid] = mx; __syncthreads();
    for (int s = 128; s > 0; s >>= 1) {
        if (tid < s) s_red[tid] = fmaxf(s_red[tid], s_red[tid + s]);
        __syncthreads();
    }
    mx = s_red[0]; __syncthreads();

    float sum = 0.f;
    for (int i = tid; i < O_; i += 256) {
        float v = __expf(row[i] - mx);
        row[i] = v;
        sum += v;
    }
    s_red[tid] = sum; __syncthreads();
    for (int s = 128; s > 0; s >>= 1) {
        if (tid < s) s_red[tid] += s_red[tid + s];
        __syncthreads();
    }
    float inv = 1.f / s_red[0];
    for (int i = tid; i < O_; i += 256) row[i] *= inv;
}

// ---------------------------------------------------------------------------
extern "C" void kernel_launch(void* const* d_in, const int* in_sizes, int n_in,
                              void* d_out, int out_size, void* d_ws, size_t ws_size,
                              hipStream_t stream) {
    (void)in_sizes; (void)n_in; (void)out_size; (void)ws_size;
    const float* inputs = (const float*)d_in[0];
    const float* stm    = (const float*)d_in[1];
    const float* x_seq  = (const float*)d_in[2];
    const float* V_a    = (const float*)d_in[3];
    const float* W_a    = (const float*)d_in[4];
    const float* W_r = (const float*)d_in[5],  *U_r = (const float*)d_in[6];
    const float* C_r = (const float*)d_in[7],  *b_r = (const float*)d_in[8];
    const float* W_z = (const float*)d_in[9],  *U_z = (const float*)d_in[10];
    const float* C_z = (const float*)d_in[11], *b_z = (const float*)d_in[12];
    const float* W_p = (const float*)d_in[13], *U_p = (const float*)d_in[14];
    const float* C_p = (const float*)d_in[15], *b_p = (const float*)d_in[16];
    const float* W_o = (const float*)d_in[17], *U_o = (const float*)d_in[18];
    const float* C_o = (const float*)d_in[19], *b_o = (const float*)d_in[20];
    float* out = (float*)d_out;

    char* ws = (char*)d_ws;
    unsigned*       WaxSw    = (unsigned*)(ws + 0);             // 262144 B
    float*          stm_part = (float*)(ws + 262144);           // 131072 B
    float*          et       = (float*)(ws + 393216);           // 524288 B
    float*          context  = (float*)(ws + 917504);           //  65536 B
    float*          rt       = (float*)(ws + 983040);           //  65536 B
    float*          zt       = (float*)(ws + 1048576);          //  65536 B
    unsigned short* act      = (unsigned short*)(ws + 1114112); //  98304 B

    k_prep_wax   <<<256,  256, 0, stream>>>(W_a, WaxSw);
    k_stm_part   <<<128,  256, 0, stream>>>(stm, W_a, stm_part);
    k_scores     <<<8192,  32, 0, stream>>>(x_seq, WaxSw, stm_part, V_a, et);
    k_softmax_ctx<<<64,   256, 0, stream>>>(et, x_seq, context);
    k_gates_rz   <<<64,   256, 0, stream>>>(inputs, stm, context,
                                            W_r, U_r, C_r, b_r,
                                            W_z, U_z, C_z, b_z, rt, zt);
    k_gates_st   <<<64,   256, 0, stream>>>(inputs, stm, context, rt, zt,
                                            W_p, U_p, C_p, b_p,
                                            out + (size_t)B_ * O_);
    k_actbf      <<<64,   256, 0, stream>>>(inputs, stm, context, act);
    k_outgemm    <<<O_/16, 32, 0, stream>>>(act, W_o, U_o, C_o, b_o, out);
    k_outsoftmax <<<64,   256, 0, stream>>>(out);
}
